// LSTM_v2_22703197127162
// MI455X (gfx1250) — compile-verified, matrix operand order checked by reading, not run on hardware
//
#include <hip/hip_runtime.h>
#include <hip/hip_bf16.h>

// ---------------------------------------------------------------------------
// 2-layer LSTM warmup (T=50) + autoregressive rollout (100 steps) on gfx1250.
// B=4096, D=64, H=128, gates G=4H=512.
//
//  * pack_weights: f32 -> bf16, pre-swizzled into per-lane WMMA B-fragment
//    order (one 32B contiguous chunk per lane per fragment).
//  * lstm_rollout: 128 WGs x 256 threads; each WG owns 32 batch rows and the
//    full 150-step recurrence with h/c state in LDS. Gate GEMMs use
//    v_wmma_f32_16x16x32_bf16 (f32 accumulation). Weight pointers are
//    laundered per step (as integers, rebuilt as address_space(1) pointers)
//    so (a) the loop-invariant weight loads cannot be hoisted/spilled and
//    (b) they lower to global_load_b128 (LOADcnt only), not flat_load
//    (which would also tie up DScnt and the LDS datapath).
// ---------------------------------------------------------------------------

typedef __attribute__((ext_vector_type(16))) __bf16 v16bf;
typedef __attribute__((ext_vector_type(8)))  __bf16 v8bf;
typedef __attribute__((ext_vector_type(8)))  float  v8f;

typedef const __attribute__((address_space(1))) __bf16* gptr_bf16;
typedef const __attribute__((address_space(1))) v16bf*  gptr_v16bf;

#define B_   4096
#define T_   50
#define D_   64
#define H_   128
#define G_   512      // 4*H
#define OUT_ 100
#define MT_  32       // batch rows per workgroup
#define NTHREADS 256  // 8 waves (wave32)

// LDS strides (elements), padded to spread banks
#define HS 136        // bf16 stride for h buffers (128 + 8)
#define XS 72         // bf16 stride for x/pred buffer (64 + 8)
#define ZS 516        // f32 stride for gate buffer (512 + 4)

// LDS layout offsets (bytes)
#define OFF_Z   0
#define OFF_C0  (OFF_Z  + MT_*ZS*4)
#define OFF_C1  (OFF_C0 + MT_*H_*4)
#define OFF_B0  (OFF_C1 + MT_*H_*4)
#define OFF_B1  (OFF_B0 + G_*4)
#define OFF_BD  (OFF_B1 + G_*4)
#define OFF_H0  (OFF_BD + 256)
#define OFF_H1  (OFF_H0 + MT_*HS*2)
#define OFF_XB  (OFF_H1 + MT_*HS*2)
#define SMEM_BYTES (OFF_XB + MT_*XS*2)   // ~125 KB

// Opaque pointer laundering -> fresh global-address-space pointer each call.
// Blocks LICM of weight loads out of the time loop (anti-spill) while keeping
// addrspace(1) so loads lower to global_load_b128 (LOADcnt only).
__device__ __forceinline__ gptr_bf16 launder_global(const __bf16* p) {
    unsigned long long v = (unsigned long long)p;
    asm volatile("" : "+s"(v));
    return (gptr_bf16)v;
}

// fast gate math on hardware exp
__device__ __forceinline__ float fast_tanh(float x) {
    return 1.f - 2.f / (__expf(2.f * x) + 1.f);
}
__device__ __forceinline__ float fast_sig(float x) {
    return 1.f / (1.f + __expf(-x));
}

// ---------------------------------------------------------------------------
// Weight packing: W is [K x N] row-major f32 -> bf16 fragments laid out
// [kt][nt][lane][16]. B fragment = A-layout of W^T (16x16x32 bf16 WMMA):
// lane l holds n = nt*16+(l&15); kb = kt*32 + ((l&16)?8:0);
// elems 0..7 -> K=kb+e, elems 8..15 -> K=kb+16+(e-8).
// ---------------------------------------------------------------------------
__global__ void pack_weights(const float* __restrict__ W, __bf16* __restrict__ out,
                             int K, int N) {
    int idx = blockIdx.x * blockDim.x + threadIdx.x;
    int total = K * N;
    if (idx >= total) return;
    int e    = idx & 15;
    int lane = (idx >> 4) & 31;
    int tile = idx >> 9;
    int NT   = N >> 4;
    int nt   = tile % NT;
    int kt   = tile / NT;
    int n    = nt * 16 + (lane & 15);
    int kb   = kt * 32 + ((lane & 16) ? 8 : 0);
    int k    = kb + ((e < 8) ? e : (8 + e));
    out[idx] = (__bf16)W[(size_t)k * N + n];
}

// ---------------------------------------------------------------------------
// Fragment loaders
// ---------------------------------------------------------------------------
__device__ __forceinline__ v16bf load_b_frag(gptr_bf16 packed, int tile, int lane) {
    return *(gptr_v16bf)(packed + ((size_t)tile * 32 + lane) * 16);
}

__device__ __forceinline__ v16bf load_a_frag_lds(const __bf16* buf, int stride,
                                                 int mbase, int kbase, int lane) {
    int row = mbase + (lane & 15);
    int k0  = kbase + ((lane & 16) ? 8 : 0);
    const __bf16* p = buf + row * stride + k0;
    v8bf lo = *reinterpret_cast<const v8bf*>(p);
    v8bf hi = *reinterpret_cast<const v8bf*>(p + 16);
    return __builtin_shufflevector(lo, hi, 0,1,2,3,4,5,6,7,8,9,10,11,12,13,14,15);
}

// ---------------------------------------------------------------------------
// Gate GEMM: z[32 x 512] = inA[32 x KIN] @ Wk + h[32 x 128] @ Wr
// Each of 8 waves owns 64 gate columns (4 N-tiles): acc = 2x4 C tiles.
// ---------------------------------------------------------------------------
template<int KIN>
__device__ __forceinline__ void gemm_gates(const __bf16* inA, int sA,
                                           const __bf16* pWkRaw,
                                           const __bf16* hbuf,
                                           const __bf16* pWrRaw,
                                           float* __restrict__ zbuf,
                                           int wave, int lane) {
    gptr_bf16 pWk = launder_global(pWkRaw);   // fresh address each timestep
    gptr_bf16 pWr = launder_global(pWrRaw);

    v8f acc[2][4];
#pragma unroll
    for (int mt = 0; mt < 2; ++mt)
#pragma unroll
        for (int j = 0; j < 4; ++j)
            acc[mt][j] = (v8f){0.f,0.f,0.f,0.f,0.f,0.f,0.f,0.f};

    const int ntBase = wave * 4;          // of 32 N-tiles total

    // input contribution (K = KIN)
#pragma unroll
    for (int kt = 0; kt < KIN / 32; ++kt) {
        v16bf a0 = load_a_frag_lds(inA, sA, 0,  kt * 32, lane);
        v16bf a1 = load_a_frag_lds(inA, sA, 16, kt * 32, lane);
#pragma unroll
        for (int j = 0; j < 4; ++j) {
            v16bf b = load_b_frag(pWk, kt * 32 + ntBase + j, lane);
            acc[0][j] = __builtin_amdgcn_wmma_f32_16x16x32_bf16(
                false, a0, false, b, (short)0, acc[0][j], false, false);
            acc[1][j] = __builtin_amdgcn_wmma_f32_16x16x32_bf16(
                false, a1, false, b, (short)0, acc[1][j], false, false);
        }
    }
    // recurrent contribution (K = 128)
#pragma unroll
    for (int kt = 0; kt < 4; ++kt) {
        v16bf a0 = load_a_frag_lds(hbuf, HS, 0,  kt * 32, lane);
        v16bf a1 = load_a_frag_lds(hbuf, HS, 16, kt * 32, lane);
#pragma unroll
        for (int j = 0; j < 4; ++j) {
            v16bf b = load_b_frag(pWr, kt * 32 + ntBase + j, lane);
            acc[0][j] = __builtin_amdgcn_wmma_f32_16x16x32_bf16(
                false, a0, false, b, (short)0, acc[0][j], false, false);
            acc[1][j] = __builtin_amdgcn_wmma_f32_16x16x32_bf16(
                false, a1, false, b, (short)0, acc[1][j], false, false);
        }
    }
    // scatter C tiles: VGPR r of lane l -> M = r + ((l&16)?8:0), N = l&15
    const int hi8 = (lane & 16) ? 8 : 0;
    const int nl  = lane & 15;
#pragma unroll
    for (int mt = 0; mt < 2; ++mt)
#pragma unroll
        for (int j = 0; j < 4; ++j) {
            int ncol = (ntBase + j) * 16 + nl;
#pragma unroll
            for (int r = 0; r < 8; ++r)
                zbuf[(mt * 16 + r + hi8) * ZS + ncol] = acc[mt][j][r];
        }
}

// ---------------------------------------------------------------------------
// Elementwise LSTM gate update (gate order i,f,g,o): 32x128 units.
// ---------------------------------------------------------------------------
__device__ __forceinline__ void lstm_elem(const float* __restrict__ zbuf,
                                          const float* __restrict__ bias,
                                          float* __restrict__ cbuf,
                                          __bf16* __restrict__ hbuf, int tid) {
    for (int u = tid; u < MT_ * H_; u += NTHREADS) {
        int m = u >> 7, j = u & 127;
        float zi = zbuf[m * ZS + j]       + bias[j];
        float zf = zbuf[m * ZS + j + 128] + bias[j + 128];
        float zg = zbuf[m * ZS + j + 256] + bias[j + 256];
        float zo = zbuf[m * ZS + j + 384] + bias[j + 384];
        float c  = fast_sig(zf) * cbuf[m * H_ + j] + fast_sig(zi) * fast_tanh(zg);
        cbuf[m * H_ + j] = c;
        hbuf[m * HS + j] = (__bf16)(fast_sig(zo) * fast_tanh(c));
    }
}

// ---------------------------------------------------------------------------
// Main rollout kernel
// ---------------------------------------------------------------------------
__global__ __launch_bounds__(NTHREADS, 4)
void lstm_rollout(const float* __restrict__ x,
                  const float* __restrict__ b0g, const float* __restrict__ b1g,
                  const float* __restrict__ bdg,
                  const __bf16* __restrict__ pWk0, const __bf16* __restrict__ pWr0,
                  const __bf16* __restrict__ pWk1, const __bf16* __restrict__ pWr1,
                  const __bf16* __restrict__ pWd,
                  float* __restrict__ out) {
    extern __shared__ char smem[];
    float*  z   = (float*)(smem + OFF_Z);
    float*  c0  = (float*)(smem + OFF_C0);
    float*  c1  = (float*)(smem + OFF_C1);
    float*  b0s = (float*)(smem + OFF_B0);
    float*  b1s = (float*)(smem + OFF_B1);
    float*  bds = (float*)(smem + OFF_BD);
    __bf16* h0  = (__bf16*)(smem + OFF_H0);
    __bf16* h1  = (__bf16*)(smem + OFF_H1);
    __bf16* xb  = (__bf16*)(smem + OFF_XB);

    const int tid   = threadIdx.x;
    const int lane  = tid & 31;
    const int wave  = tid >> 5;
    const int brow0 = blockIdx.x * MT_;

    for (int u = tid; u < G_; u += NTHREADS) { b0s[u] = b0g[u]; b1s[u] = b1g[u]; }
    if (tid < D_) bds[tid] = bdg[tid];
    for (int u = tid; u < MT_ * H_; u += NTHREADS) { c0[u] = 0.f; c1[u] = 0.f; }
    for (int u = tid; u < MT_ * HS; u += NTHREADS) { h0[u] = (__bf16)0.f; h1[u] = (__bf16)0.f; }
    __syncthreads();

    // ---- warmup over the observed sequence ----
    for (int t = 0; t < T_; ++t) {
        for (int u = tid; u < MT_ * D_; u += NTHREADS) {
            int m = u >> 6, d = u & 63;
            xb[m * XS + d] = (__bf16)x[((size_t)(brow0 + m) * T_ + t) * D_ + d];
        }
        __syncthreads();
        gemm_gates<D_>(xb, XS, pWk0, h0, pWr0, z, wave, lane);
        __syncthreads();
        lstm_elem(z, b0s, c0, h0, tid);
        __syncthreads();
        gemm_gates<H_>(h0, HS, pWk1, h1, pWr1, z, wave, lane);
        __syncthreads();
        lstm_elem(z, b1s, c1, h1, tid);
        __syncthreads();
    }

    // ---- autoregressive rollout ----
    for (int s = 0; s < OUT_; ++s) {
        if (s > 0) {
            gemm_gates<D_>(xb, XS, pWk0, h0, pWr0, z, wave, lane);
            __syncthreads();
            lstm_elem(z, b0s, c0, h0, tid);
            __syncthreads();
            gemm_gates<H_>(h0, HS, pWk1, h1, pWr1, z, wave, lane);
            __syncthreads();
            lstm_elem(z, b1s, c1, h1, tid);
            __syncthreads();
        }
        // dense: pred[32x64] = h1 @ Wd + bd ; one 16x16 tile per wave
        {
            gptr_bf16 pWdL = launder_global(pWd);
            const int mt = wave >> 2, nt = wave & 3;
            v8f acc = (v8f){0.f,0.f,0.f,0.f,0.f,0.f,0.f,0.f};
#pragma unroll
            for (int kt = 0; kt < 4; ++kt) {
                v16bf a = load_a_frag_lds(h1, HS, mt * 16, kt * 32, lane);
                v16bf b = load_b_frag(pWdL, kt * 4 + nt, lane);
                acc = __builtin_amdgcn_wmma_f32_16x16x32_bf16(
                    false, a, false, b, (short)0, acc, false, false);
            }
            const int hi8 = (lane & 16) ? 8 : 0;
            const int nl  = lane & 15;
            const int n   = nt * 16 + nl;
#pragma unroll
            for (int r = 0; r < 8; ++r) {
                int m = mt * 16 + r + hi8;
                float v = acc[r] + bds[n];
                out[((size_t)(brow0 + m) * OUT_ + s) * D_ + n] = v;
                xb[m * XS + n] = (__bf16)v;   // feedback input for next step
            }
        }
        __syncthreads();
    }
}

// ---------------------------------------------------------------------------
// Launch
// ---------------------------------------------------------------------------
extern "C" void kernel_launch(void* const* d_in, const int* in_sizes, int n_in,
                              void* d_out, int out_size, void* d_ws, size_t ws_size,
                              hipStream_t stream) {
    const float* x   = (const float*)d_in[0];
    const float* Wk0 = (const float*)d_in[1];
    const float* Wr0 = (const float*)d_in[2];
    const float* b0  = (const float*)d_in[3];
    const float* Wk1 = (const float*)d_in[4];
    const float* Wr1 = (const float*)d_in[5];
    const float* b1  = (const float*)d_in[6];
    const float* Wd  = (const float*)d_in[7];
    const float* bd  = (const float*)d_in[8];
    float* out = (float*)d_out;

    char* ws = (char*)d_ws;
    __bf16* pWk0 = (__bf16*)(ws);                         // 64 KB
    __bf16* pWr0 = (__bf16*)(ws + 65536);                 // 128 KB
    __bf16* pWk1 = (__bf16*)(ws + 65536 + 131072);
    __bf16* pWr1 = (__bf16*)(ws + 65536 + 2 * 131072);
    __bf16* pWd  = (__bf16*)(ws + 65536 + 3 * 131072);    // 16 KB

    pack_weights<<<(D_ * G_ + 255) / 256, 256, 0, stream>>>(Wk0, pWk0, D_, G_);
    pack_weights<<<(H_ * G_ + 255) / 256, 256, 0, stream>>>(Wr0, pWr0, H_, G_);
    pack_weights<<<(H_ * G_ + 255) / 256, 256, 0, stream>>>(Wk1, pWk1, H_, G_);
    pack_weights<<<(H_ * G_ + 255) / 256, 256, 0, stream>>>(Wr1, pWr1, H_, G_);
    pack_weights<<<(H_ * D_ + 255) / 256, 256, 0, stream>>>(Wd,  pWd,  H_, D_);

    lstm_rollout<<<B_ / MT_, NTHREADS, SMEM_BYTES, stream>>>(
        x, b0, b1, bd, pWk0, pWr0, pWk1, pWr1, pWd, out);
}
